// OriginalDefDetr_11776800325997
// MI455X (gfx1250) — compile-verified, matrix operand order checked by reading, not run on hardware
//
#include <hip/hip_runtime.h>

// ---------------- model constants ----------------
#define BATCH   2
#define DIM     256
#define NHEAD   8
#define DHEAD   32
#define NLVL    4
#define NPT     4
#define DFF     1024
#define NQD     300
#define S_TOT   11253           // 92^2 + 46^2 + 23^2 + 12^2
#define MP      22528           // pad(B*S_TOT=22506) to 128
#define MPD     640             // pad(B*NQD=600)   to 128
#define SCALE_ATT 0.17677669529663687f   // 1/sqrt(32)

typedef _Float16 vh16 __attribute__((ext_vector_type(16)));
typedef _Float16 vh8  __attribute__((ext_vector_type(8)));
typedef _Float16 vh4  __attribute__((ext_vector_type(4)));
typedef float    vf8  __attribute__((ext_vector_type(8)));
typedef float    vf4  __attribute__((ext_vector_type(4)));

union FragH { vh16 v; unsigned u[8]; };
union AccU  { vf8  v; float f[8]; };

// =====================================================================
// Tiled WMMA GEMM with 2-stage LDS ping-pong:
//   C[M,N] = A[M,K] @ W[K,N] + bias   (A: f16 row-major, W pre-transposed
//   f16 [N][K] row-major, C: f32 or f16, optional ReLU)
// M,N multiples of 128, K multiple of 32. 256 threads = 8 waves;
// wave (wr 0..3, wc 0..1) computes 32x64 = 2x4 v_wmma_f32_16x16x32_f16.
// =====================================================================
#define BK 32
#define LDSST 40   // halfs per LDS row (pad to dodge bank conflicts)

template<int RELU, int OUTHALF>
__global__ __launch_bounds__(256) void gemm_wmma_f16(
    const _Float16* __restrict__ A, const _Float16* __restrict__ Wt,
    const float* __restrict__ bias, void* __restrict__ Cout,
    int M, int N, int K)
{
  __shared__ _Float16 sA[2][128 * LDSST];
  __shared__ _Float16 sB[2][128 * LDSST];

  const int tid  = threadIdx.x;
  const int lane = tid & 31;
  const int wid  = tid >> 5;
  const int wr   = wid & 3;        // 4 row-groups of 32 rows
  const int wc   = wid >> 2;       // 2 col-groups of 64 cols
  const int tileM = blockIdx.y * 128;
  const int tileN = blockIdx.x * 128;

  vf8 acc[2][4];
  const vf8 vzero = {0.f,0.f,0.f,0.f,0.f,0.f,0.f,0.f};
#pragma unroll
  for (int i = 0; i < 2; ++i)
#pragma unroll
    for (int j = 0; j < 4; ++j) acc[i][j] = vzero;

  const int khalf = (lane >> 4) << 3;   // 0 or 8 (16-bit fragment layout)
  const int fidx  = lane & 15;          // row (A) / col (B) within 16

  // staging registers: 2 x 16B per matrix per thread covers a 128x32 tile
  vh8 ra[2], rb[2];
  auto loadRegs = [&](int kt) {
#pragma unroll
    for (int i = 0; i < 2; ++i) {
      int e   = tid + (i << 8);        // 0..511
      int row = e >> 2;                // 0..127
      int col = (e & 3) << 3;          // 0,8,16,24
      ra[i] = *(const vh8*)(A  + (size_t)(tileM + row) * K + kt + col);
      rb[i] = *(const vh8*)(Wt + (size_t)(tileN + row) * K + kt + col);
    }
  };
  auto storeRegs = [&](int st) {
#pragma unroll
    for (int i = 0; i < 2; ++i) {
      int e   = tid + (i << 8);
      int row = e >> 2;
      int col = (e & 3) << 3;
      *(vh8*)&sA[st][row * LDSST + col] = ra[i];
      *(vh8*)&sB[st][row * LDSST + col] = rb[i];
    }
  };

  loadRegs(0);
  storeRegs(0);
  int st = 0;
  for (int kt = 0; kt < K; kt += BK) {
    __syncthreads();
    const bool more = (kt + BK) < K;
    if (more) loadRegs(kt + BK);           // overlap global loads with WMMA

    const _Float16* pA = sA[st];
    const _Float16* pB = sB[st];
    FragH af[2], bf[4];
#pragma unroll
    for (int rt = 0; rt < 2; ++rt) {
      const _Float16* base = pA + (wr * 32 + rt * 16 + fidx) * LDSST;
#pragma unroll
      for (int v = 0; v < 8; ++v) {
        int kb = ((v >> 2) << 4) + khalf + ((v & 3) << 1);
        af[rt].u[v] = *(const unsigned*)(base + kb);
      }
    }
#pragma unroll
    for (int ct = 0; ct < 4; ++ct) {
      const _Float16* base = pB + (wc * 64 + ct * 16 + fidx) * LDSST;
#pragma unroll
      for (int v = 0; v < 8; ++v) {
        int kb = ((v >> 2) << 4) + khalf + ((v & 3) << 1);
        bf[ct].u[v] = *(const unsigned*)(base + kb);
      }
    }
#pragma unroll
    for (int rt = 0; rt < 2; ++rt)
#pragma unroll
      for (int ct = 0; ct < 4; ++ct)
        acc[rt][ct] = __builtin_amdgcn_wmma_f32_16x16x32_f16(
            false, af[rt].v, false, bf[ct].v, (short)0, acc[rt][ct], false, false);

    if (more) storeRegs(st ^ 1);
    st ^= 1;
  }

  // ---- epilogue: bias (+relu), f32 or f16 store ----
  const int msub = (lane >> 4) << 3;   // C/D layout: lanes16-31 hold M+8
  const int ncol = lane & 15;
#pragma unroll
  for (int rt = 0; rt < 2; ++rt) {
#pragma unroll
    for (int ct = 0; ct < 4; ++ct) {
      const int n = tileN + wc * 64 + ct * 16 + ncol;
      const float bv = bias[n];
      AccU au; au.v = acc[rt][ct];
#pragma unroll
      for (int r = 0; r < 8; ++r) {
        int m = tileM + wr * 32 + rt * 16 + msub + r;
        float val = au.f[r] + bv;
        if (RELU) val = fmaxf(val, 0.f);
        if (OUTHALF) ((_Float16*)Cout)[(size_t)m * N + n] = (_Float16)val;
        else         ((float*)Cout)[(size_t)m * N + n] = val;
      }
    }
  }
}

// =====================================================================
// weight pre-convert + pre-transpose: f32 [K][N] -> f16 [N][K]
// one kernel per layer (region if-chain over concatenated block)
// =====================================================================
#define EW_VAL 0
#define EW_OFF 65536
#define EW_AW  131072
#define EW_OUT 163840
#define EW_F1  229376
#define EW_F2  491520
#define EW_TOT 753664
__global__ void convw_enc(const float* __restrict__ v, const float* __restrict__ o,
                          const float* __restrict__ a, const float* __restrict__ ot,
                          const float* __restrict__ f1, const float* __restrict__ f2,
                          _Float16* __restrict__ dst) {
  int idx = blockIdx.x * blockDim.x + threadIdx.x;
  if (idx >= EW_TOT) return;
  const float* src; int base, K, N;
  if      (idx < EW_OFF) { src = v;  base = EW_VAL; K = 256;  N = 256;  }
  else if (idx < EW_AW)  { src = o;  base = EW_OFF; K = 256;  N = 256;  }
  else if (idx < EW_OUT) { src = a;  base = EW_AW;  K = 256;  N = 128;  }
  else if (idx < EW_F1)  { src = ot; base = EW_OUT; K = 256;  N = 256;  }
  else if (idx < EW_F2)  { src = f1; base = EW_F1;  K = 256;  N = 1024; }
  else                   { src = f2; base = EW_F2;  K = 1024; N = 256;  }
  int e = idx - base; int n = e / K; int k = e - n * K;
  dst[idx] = (_Float16)src[(size_t)k * N + n];
}

#define DW_Q     0
#define DW_K     65536
#define DW_V     131072
#define DW_O     196608
#define DW_CAV   262144
#define DW_CAO   327680
#define DW_CAOUT 393216
#define DW_CAAW  458752
#define DW_F1    491520
#define DW_F2    753664
#define DW_TOT   983040
__global__ void convw_dec(const float* __restrict__ q, const float* __restrict__ k_,
                          const float* __restrict__ v, const float* __restrict__ o,
                          const float* __restrict__ cav, const float* __restrict__ cao,
                          const float* __restrict__ caout, const float* __restrict__ caaw,
                          const float* __restrict__ f1, const float* __restrict__ f2,
                          _Float16* __restrict__ dst) {
  int idx = blockIdx.x * blockDim.x + threadIdx.x;
  if (idx >= DW_TOT) return;
  const float* src; int base, K, N;
  if      (idx < DW_K)     { src = q;     base = DW_Q;     K = 256;  N = 256;  }
  else if (idx < DW_V)     { src = k_;    base = DW_K;     K = 256;  N = 256;  }
  else if (idx < DW_O)     { src = v;     base = DW_V;     K = 256;  N = 256;  }
  else if (idx < DW_CAV)   { src = o;     base = DW_O;     K = 256;  N = 256;  }
  else if (idx < DW_CAO)   { src = cav;   base = DW_CAV;   K = 256;  N = 256;  }
  else if (idx < DW_CAOUT) { src = cao;   base = DW_CAO;   K = 256;  N = 256;  }
  else if (idx < DW_CAAW)  { src = caout; base = DW_CAOUT; K = 256;  N = 256;  }
  else if (idx < DW_F1)    { src = caaw;  base = DW_CAAW;  K = 256;  N = 128;  }
  else if (idx < DW_F2)    { src = f1;    base = DW_F1;    K = 256;  N = 1024; }
  else                     { src = f2;    base = DW_F2;    K = 1024; N = 256;  }
  int e = idx - base; int n = e / K; int k = e - n * K;
  dst[idx] = (_Float16)src[(size_t)k * N + n];
}

// =====================================================================
// elementwise / packing / attention kernels
// =====================================================================
__global__ void pack_kernel(const float* __restrict__ src, float* __restrict__ X,
                            _Float16* __restrict__ X16, int HW, int soff) {
  int idx = blockIdx.x * blockDim.x + threadIdx.x;
  int total = BATCH * HW * DIM;
  if (idx >= total) return;
  int d = idx & 255; int t = idx >> 8;
  int p = t % HW;    int b = t / HW;
  float v = src[(size_t)(b * DIM + d) * HW + p];
  size_t oi = (size_t)(b * S_TOT + soff + p) * DIM + d;
  X[oi] = v; X16[oi] = (_Float16)v;
}

__global__ void ref_kernel(float* __restrict__ REF) {
  int s = blockIdx.x * blockDim.x + threadIdx.x;
  if (s >= S_TOT) return;
  int off, Wl, Hl;
  if      (s <  8464) { off = 0;     Wl = 92; Hl = 92; }
  else if (s < 10580) { off = 8464;  Wl = 46; Hl = 46; }
  else if (s < 11109) { off = 10580; Wl = 23; Hl = 23; }
  else                { off = 11109; Wl = 12; Hl = 12; }
  int p = s - off; int y = p / Wl; int x = p % Wl;
  REF[s * 2 + 0] = (x + 0.5f) / (float)Wl;
  REF[s * 2 + 1] = (y + 0.5f) / (float)Hl;
}

__global__ void addpos_kernel(const float* __restrict__ X, const float* __restrict__ lvl,
                              _Float16* __restrict__ XP16) {
  int idx = blockIdx.x * blockDim.x + threadIdx.x;
  int total = BATCH * S_TOT * (DIM / 4);
  if (idx >= total) return;
  int d4 = (idx & 63) << 2; int t = idx >> 6;
  int s = t % S_TOT; int b = t / S_TOT;
  int l = (s < 8464) ? 0 : (s < 10580) ? 1 : (s < 11109) ? 2 : 3;
  vf4 xv = *(const vf4*)(X   + (size_t)(b * S_TOT + s) * DIM + d4);
  vf4 pv = *(const vf4*)(lvl + (size_t)l * DIM + d4);
  vh4 h;
#pragma unroll
  for (int i = 0; i < 4; ++i) h[i] = (_Float16)(xv[i] + pv[i]);
  *(vh4*)(XP16 + (size_t)(b * S_TOT + s) * DIM + d4) = h;
}

__global__ void add_h_kernel(const float* __restrict__ A, const float* __restrict__ Bp,
                             _Float16* __restrict__ C, int n4) {
  int idx = blockIdx.x * blockDim.x + threadIdx.x;
  if (idx >= n4) return;
  vf4 a = *(const vf4*)(A  + (size_t)idx * 4);
  vf4 b = *(const vf4*)(Bp + (size_t)idx * 4);
  vh4 h;
#pragma unroll
  for (int i = 0; i < 4; ++i) h[i] = (_Float16)(a[i] + b[i]);
  *(vh4*)(C + (size_t)idx * 4) = h;
}

__global__ void softmax16_kernel(float* __restrict__ AW, int rows) {
  int idx = blockIdx.x * blockDim.x + threadIdx.x;
  if (idx >= rows * NHEAD) return;
  int h = idx % NHEAD; int r = idx / NHEAD;
  float* p = AW + (size_t)r * (NHEAD * 16) + h * 16;
  float mx = -3.4e38f;
#pragma unroll
  for (int i = 0; i < 16; ++i) mx = fmaxf(mx, p[i]);
  float s = 0.f; float e[16];
#pragma unroll
  for (int i = 0; i < 16; ++i) { e[i] = __expf(p[i] - mx); s += e[i]; }
  float inv = 1.f / s;
#pragma unroll
  for (int i = 0; i < 16; ++i) p[i] = e[i] * inv;
}

// one wave per (b, q, head); lane = channel within head (Dh = 32)
__global__ void msdeform_kernel(const float* __restrict__ VAL,
                                const float* __restrict__ OFF,
                                const float* __restrict__ AW,
                                const float* __restrict__ REF,
                                _Float16* __restrict__ OUT16, int Lq) {
  const int Hs[4] = {92, 46, 23, 12};
  const int Ws[4] = {92, 46, 23, 12};
  const int st[4] = {0, 8464, 10580, 11109};
  int gw = blockIdx.x * 8 + (threadIdx.x >> 5);
  int lane = threadIdx.x & 31;
  if (gw >= BATCH * Lq * NHEAD) return;
  int h = gw % NHEAD; int t = gw / NHEAD;
  int q = t % Lq;     int b = t / Lq;
  const size_t rowm = (size_t)b * Lq + q;
  const float refx = REF[q * 2 + 0], refy = REF[q * 2 + 1];
  const float* offR = OFF + rowm * DIM + h * (NLVL * NPT * 2);
  const float* awR  = AW  + rowm * (NHEAD * 16) + h * 16;
  const size_t vbase = (size_t)b * S_TOT * DIM + h * DHEAD + lane;
  float acc = 0.f;
#pragma unroll
  for (int l = 0; l < NLVL; ++l) {
    const int Wl = Ws[l], Hl = Hs[l], base = st[l];
    const float fW = (float)Wl, fH = (float)Hl;
#pragma unroll
    for (int p = 0; p < NPT; ++p) {
      int ip = l * NPT + p;
      float ox = offR[ip * 2 + 0], oy = offR[ip * 2 + 1];
      float a  = awR[ip];
      float x = (refx + ox / fW) * fW - 0.5f;
      float y = (refy + oy / fH) * fH - 0.5f;
      float fx0 = floorf(x), fy0 = floorf(y);
      int x0 = (int)fx0, y0 = (int)fy0;
      float fx = x - fx0, fy = y - fy0;
#pragma unroll
      for (int c = 0; c < 4; ++c) {
        int xi = x0 + (c & 1), yi = y0 + (c >> 1);
        float wgt = ((c & 1) ? fx : 1.f - fx) * ((c >> 1) ? fy : 1.f - fy);
        if (xi >= 0 && xi < Wl && yi >= 0 && yi < Hl) {
          acc += a * wgt * VAL[vbase + (size_t)(base + yi * Wl + xi) * DIM];
        }
      }
    }
  }
  OUT16[rowm * DIM + h * DHEAD + lane] = (_Float16)acc;
}

// one wave per row: y = LN(x + r) * g + b ; writes f32 and f16 mirrors
__global__ void ln_res_kernel(const float* __restrict__ X, const float* __restrict__ R,
                              const float* __restrict__ g, const float* __restrict__ bt,
                              float* __restrict__ Y, _Float16* __restrict__ Y16, int rows) {
  int row  = blockIdx.x * 8 + (threadIdx.x >> 5);
  int lane = threadIdx.x & 31;
  if (row >= rows) return;
  const float* xr = X + (size_t)row * DIM;
  const float* rr = R + (size_t)row * DIM;
  float v[8]; float s = 0.f;
#pragma unroll
  for (int i = 0; i < 8; ++i) { v[i] = xr[lane + i * 32] + rr[lane + i * 32]; s += v[i]; }
#pragma unroll
  for (int o = 16; o > 0; o >>= 1) s += __shfl_xor(s, o, 32);
  float mean = s * (1.f / DIM);
  float vs = 0.f;
#pragma unroll
  for (int i = 0; i < 8; ++i) { float d = v[i] - mean; vs += d * d; }
#pragma unroll
  for (int o = 16; o > 0; o >>= 1) vs += __shfl_xor(vs, o, 32);
  float inv = rsqrtf(vs * (1.f / DIM) + 1e-5f);
#pragma unroll
  for (int i = 0; i < 8; ++i) {
    int c = lane + i * 32;
    float yv = (v[i] - mean) * inv * g[c] + bt[c];
    Y  [(size_t)row * DIM + c] = yv;
    Y16[(size_t)row * DIM + c] = (_Float16)yv;
  }
}

// decoder self-attention: one wave per (b, h, q), lane = channel, online softmax
__global__ void mha_kernel(const float* __restrict__ QH, const float* __restrict__ KH,
                           const float* __restrict__ VH, const unsigned char* __restrict__ mask,
                           _Float16* __restrict__ OUT16) {
  int gw = blockIdx.x * 8 + (threadIdx.x >> 5);
  int lane = threadIdx.x & 31;
  if (gw >= BATCH * NHEAD * NQD) return;
  int q = gw % NQD; int t = gw / NQD;
  int h = t % NHEAD; int b = t / NHEAD;
  const size_t rq = (size_t)(b * NQD + q) * DIM + h * DHEAD + lane;
  const float qv = QH[rq];
  float m = -3.0e38f, l = 0.f, acc = 0.f;
  for (int k = 0; k < NQD; ++k) {
    const size_t rk = (size_t)(b * NQD + k) * DIM + h * DHEAD + lane;
    float prod = qv * KH[rk];
#pragma unroll
    for (int o = 16; o > 0; o >>= 1) prod += __shfl_xor(prod, o, 32);
    float s = prod * SCALE_ATT;
    if (mask[q * NQD + k]) s = -1e9f;
    float mn = fmaxf(m, s);
    float alpha = __expf(m - mn);
    float p = __expf(s - mn);
    acc = acc * alpha + p * VH[rk];
    l = l * alpha + p;
    m = mn;
  }
  OUT16[rq] = (_Float16)(acc / l);
}

__global__ void bcast_kernel(const float* __restrict__ qe, float* __restrict__ TGT,
                             _Float16* __restrict__ TGT16, float* __restrict__ QPOS) {
  int idx = blockIdx.x * blockDim.x + threadIdx.x;
  if (idx >= NQD * DIM) return;
  int d = idx & 255; int q = idx >> 8;
  float tp = qe[(size_t)q * 512 + d];
  float tt = qe[(size_t)q * 512 + DIM + d];
#pragma unroll
  for (int b = 0; b < BATCH; ++b) {
    size_t oi = (size_t)(b * NQD + q) * DIM + d;
    TGT[oi] = tt; TGT16[oi] = (_Float16)tt; QPOS[oi] = tp;
  }
}

__global__ void sigmoid_kernel(const float* __restrict__ qr, float* __restrict__ DREF) {
  int idx = blockIdx.x * blockDim.x + threadIdx.x;
  if (idx >= NQD * 2) return;
  DREF[idx] = 1.f / (1.f + __expf(-qr[idx]));
}

// =====================================================================
// host orchestration
// =====================================================================
static void gemm(const _Float16* A, const _Float16* Wt, const float* b, void* C,
                 int M, int N, int K, int relu_half, hipStream_t s) {
  dim3 grid(N / 128, M / 128), blk(256);
  if (relu_half) gemm_wmma_f16<1, 1><<<grid, blk, 0, s>>>(A, Wt, b, C, M, N, K);
  else           gemm_wmma_f16<0, 0><<<grid, blk, 0, s>>>(A, Wt, b, C, M, N, K);
}

extern "C" void kernel_launch(void* const* d_in, const int* in_sizes, int n_in,
                              void* d_out, int out_size, void* d_ws, size_t ws_size,
                              hipStream_t stream) {
  (void)in_sizes; (void)n_in; (void)out_size; (void)ws_size;
  auto F = [&](int i) { return (const float*)d_in[i]; };

  // input order: src0..3, q_emb, q_ref, attn_mask, then jax-pytree params
  // (sorted keys): dec[6]{ca{aw,off,out,value},f1,f2,k,n1,n2,n3,o,q,v},
  //                enc[6]{f1,f2,n1,n2,sa{aw,off,out,value}}, lvl
  const int PB = 7;
  auto DB = [&](int i) { return PB + i * 26; };
  auto EB = [&](int i) { return PB + 156 + i * 16; };
  const int LVL = PB + 156 + 96;

  // ---------- workspace carve: f32 region then f16 region ----------
  float* ws = (float*)d_ws;
  const size_t SZ  = (size_t)MP * DIM;
  const size_t SZD = (size_t)MPD * DIM;
  float* X    = ws;
  float* XP   = X   + SZ;          // encoder O-proj / scratch (f32)
  float* VAL  = XP  + SZ;
  float* OFF  = VAL + SZ;          // also FFN-down destination
  float* AW   = OFF + SZ;          // MP*128
  float* TGT  = AW  + (size_t)MP * 128;
  float* QPOS = TGT  + SZD;
  float* QH   = QPOS + SZD;
  float* KH   = QH   + SZD;
  float* VH   = KH   + SZD;
  float* OPRJ = VH   + SZD;
  float* DOFF = OPRJ + SZD;
  float* DAW  = DOFF + SZD;        // MPD*128
  float* REF2 = DAW  + (size_t)MPD * 128;
  float* DREF = REF2 + 2 * ((size_t)S_TOT + 11);   // padded to keep alignment
  float* FEND = DREF + 2 * (size_t)NQD + 40;

  _Float16* hb = (_Float16*)FEND;
  _Float16* X16    = hb;
  _Float16* XP16   = X16    + SZ;
  _Float16* SOUT16 = XP16   + SZ;
  _Float16* F116   = SOUT16 + SZ;            // MP*1024
  _Float16* TGT16  = F116   + (size_t)MP * DFF;
  _Float16* QBUF16 = TGT16  + SZD;
  _Float16* ATT16  = QBUF16 + SZD;
  _Float16* DSAMP16= ATT16  + SZD;
  _Float16* WENC   = DSAMP16+ SZD;           // 6 * EW_TOT
  _Float16* WDEC   = WENC   + 6 * (size_t)EW_TOT;

  const unsigned char* mask = (const unsigned char*)d_in[6];
  const int rowsE = BATCH * S_TOT;   // 22506
  const int rowsD = BATCH * NQD;     // 600

  // ---- setup: pack inputs, refs, weight pre-convert/transpose ----
  {
    const int HWs[4]  = {92 * 92, 46 * 46, 23 * 23, 12 * 12};
    const int offs[4] = {0, 8464, 10580, 11109};
    for (int l = 0; l < 4; ++l) {
      int tot = BATCH * HWs[l] * DIM;
      pack_kernel<<<(tot + 255) / 256, 256, 0, stream>>>(F(l), X, X16, HWs[l], offs[l]);
    }
    ref_kernel<<<(S_TOT + 255) / 256, 256, 0, stream>>>(REF2);
    bcast_kernel<<<(NQD * DIM + 255) / 256, 256, 0, stream>>>(F(4), TGT, TGT16, QPOS);
    sigmoid_kernel<<<(NQD * 2 + 255) / 256, 256, 0, stream>>>(F(5), DREF);
    for (int i = 0; i < 6; ++i) {
      const int e = EB(i);
      convw_enc<<<(EW_TOT + 255) / 256, 256, 0, stream>>>(
          F(e + 14), F(e + 10), F(e + 8), F(e + 12), F(e + 0), F(e + 2),
          WENC + (size_t)i * EW_TOT);
      const int d = DB(i);
      convw_dec<<<(DW_TOT + 255) / 256, 256, 0, stream>>>(
          F(d + 22), F(d + 12), F(d + 24), F(d + 20), F(d + 6), F(d + 2),
          F(d + 4), F(d + 0), F(d + 8), F(d + 10),
          WDEC + (size_t)i * DW_TOT);
    }
  }

  // ---- encoder: 6 layers ----
  for (int i = 0; i < 6; ++i) {
    const int e = EB(i);
    const _Float16* wb = WENC + (size_t)i * EW_TOT;
    addpos_kernel<<<(BATCH * S_TOT * 64 + 255) / 256, 256, 0, stream>>>(X, F(LVL), XP16);
    gemm(X16,  wb + EW_VAL, F(e + 15), VAL, MP, 256, 256, 0, stream);
    gemm(XP16, wb + EW_OFF, F(e + 11), OFF, MP, 256, 256, 0, stream);
    gemm(XP16, wb + EW_AW,  F(e + 9),  AW,  MP, 128, 256, 0, stream);
    softmax16_kernel<<<(rowsE * NHEAD + 255) / 256, 256, 0, stream>>>(AW, rowsE);
    msdeform_kernel<<<(rowsE * NHEAD + 7) / 8, 256, 0, stream>>>(VAL, OFF, AW, REF2, SOUT16, S_TOT);
    gemm(SOUT16, wb + EW_OUT, F(e + 13), XP, MP, 256, 256, 0, stream);
    ln_res_kernel<<<(rowsE + 7) / 8, 256, 0, stream>>>(X, XP, F(e + 4), F(e + 5), X, X16, rowsE);
    gemm(X16,  wb + EW_F1, F(e + 1), F116, MP, DFF, 256, 1, stream);   // relu, f16 out
    gemm(F116, wb + EW_F2, F(e + 3), OFF,  MP, 256, DFF, 0, stream);
    ln_res_kernel<<<(rowsE + 7) / 8, 256, 0, stream>>>(X, OFF, F(e + 6), F(e + 7), X, X16, rowsE);
  }

  // ---- decoder: 6 layers (memory = X / X16) ----
  for (int i = 0; i < 6; ++i) {
    const int d = DB(i);
    const _Float16* wb = WDEC + (size_t)i * DW_TOT;
    add_h_kernel<<<(rowsD * 64 + 255) / 256, 256, 0, stream>>>(TGT, QPOS, QBUF16, rowsD * 64);
    gemm(QBUF16, wb + DW_Q, F(d + 23), QH, MPD, 256, 256, 0, stream);
    gemm(QBUF16, wb + DW_K, F(d + 13), KH, MPD, 256, 256, 0, stream);
    gemm(TGT16,  wb + DW_V, F(d + 25), VH, MPD, 256, 256, 0, stream);
    mha_kernel<<<(BATCH * NHEAD * NQD + 7) / 8, 256, 0, stream>>>(QH, KH, VH, mask, ATT16);
    gemm(ATT16, wb + DW_O, F(d + 21), OPRJ, MPD, 256, 256, 0, stream);
    ln_res_kernel<<<(rowsD + 7) / 8, 256, 0, stream>>>(TGT, OPRJ, F(d + 16), F(d + 17), TGT, TGT16, rowsD);
    add_h_kernel<<<(rowsD * 64 + 255) / 256, 256, 0, stream>>>(TGT, QPOS, QBUF16, rowsD * 64);
    gemm(X16,    wb + DW_CAV,  F(d + 7), VAL,  MP,  256, 256, 0, stream);   // value over memory
    gemm(QBUF16, wb + DW_CAO,  F(d + 3), DOFF, MPD, 256, 256, 0, stream);
    gemm(QBUF16, wb + DW_CAAW, F(d + 1), DAW,  MPD, 128, 256, 0, stream);
    softmax16_kernel<<<(rowsD * NHEAD + 255) / 256, 256, 0, stream>>>(DAW, rowsD);
    msdeform_kernel<<<(rowsD * NHEAD + 7) / 8, 256, 0, stream>>>(VAL, DOFF, DAW, DREF, DSAMP16, NQD);
    gemm(DSAMP16, wb + DW_CAOUT, F(d + 5), OPRJ, MPD, 256, 256, 0, stream);
    ln_res_kernel<<<(rowsD + 7) / 8, 256, 0, stream>>>(TGT, OPRJ, F(d + 14), F(d + 15), TGT, TGT16, rowsD);
    gemm(TGT16, wb + DW_F1, F(d + 9),  F116, MPD, DFF, 256, 1, stream);     // relu, f16 out
    gemm(F116,  wb + DW_F2, F(d + 11), OPRJ, MPD, 256, DFF, 0, stream);
    ln_res_kernel<<<(rowsD + 7) / 8, 256, 0, stream>>>(TGT, OPRJ, F(d + 18), F(d + 19), TGT, TGT16, rowsD);
    hipMemcpyAsync((float*)d_out + (size_t)i * rowsD * DIM, TGT,
                   (size_t)rowsD * DIM * sizeof(float), hipMemcpyDeviceToDevice, stream);
  }
}
